// post_unet_38362647888311
// MI455X (gfx1250) — matrix operand order repaired and checked by reading, not compiled, not run on hardware
//
#include <hip/hip_runtime.h>
#include <cstdint>

// ---------------------------------------------------------------------------
// CDNA5 (gfx1250) WMMA types
// ---------------------------------------------------------------------------
typedef __bf16 v16bf __attribute__((ext_vector_type(16)));
typedef float  v8f   __attribute__((ext_vector_type(8)));

union Frag32B { uint4 q[2]; unsigned short u[16]; v16bf v; };

__device__ inline unsigned short f2bf(float f) {
  unsigned int x = __float_as_uint(f);
  unsigned int r = (x + 0x7fffu + ((x >> 16) & 1u)) >> 16;   // RNE
  return (unsigned short)r;
}

// Hot-path A-fragment build: native v_cvt bf16 converts (RNE), packed by clang.
__device__ inline void loadA(Frag32B& a, const float* p) {
  const float4 f0 = *(const float4*)(p);
  const float4 f1 = *(const float4*)(p + 4);
  const float4 f2 = *(const float4*)(p + 8);
  const float4 f3 = *(const float4*)(p + 12);
  a.v[0]  = (__bf16)f0.x; a.v[1]  = (__bf16)f0.y; a.v[2]  = (__bf16)f0.z; a.v[3]  = (__bf16)f0.w;
  a.v[4]  = (__bf16)f1.x; a.v[5]  = (__bf16)f1.y; a.v[6]  = (__bf16)f1.z; a.v[7]  = (__bf16)f1.w;
  a.v[8]  = (__bf16)f2.x; a.v[9]  = (__bf16)f2.y; a.v[10] = (__bf16)f2.z; a.v[11] = (__bf16)f2.w;
  a.v[12] = (__bf16)f3.x; a.v[13] = (__bf16)f3.y; a.v[14] = (__bf16)f3.z; a.v[15] = (__bf16)f3.w;
}

// ---------------------------------------------------------------------------
// Generic implicit-GEMM conv kernel on v_wmma_f32_16x16x32_bf16.
//   X   : NHWC f32 activations (Cin % 32 == 0)
//   Wp  : prepacked bf16 weights, layout pb[(kc*Npad + n)*32 + kk],
//         k = kc*32+kk = (ky*KW+kx)*Cin + c
//   Out : NHWC f32 (row stride Cstride, channel offset coff), or NCHW,
//         optionally 2x upsample-scatter (transpose conv slice).
// One wave computes one 16x16 C tile; 8 waves / block. No divisions in the
// hot loop: (ky,kx) unroll via template, channel chunk is a strided pointer.
// act: 0=none 1=relu 2=sigmoid
// ---------------------------------------------------------------------------
template<int KH, int KW>
__global__ __launch_bounds__(256) void conv_wmma_k(
    const float* __restrict__ X, const unsigned short* __restrict__ Wp,
    const float* __restrict__ scale, const float* __restrict__ bias,
    float* __restrict__ Out,
    int Nimg, int H, int W, int Cin,
    int Nout, int Npad,
    int ups, int ua, int ub,
    int Cstride, int coff, int nchw, int act)
{
  const int HW = H * W;
  const int M  = Nimg * HW;
  const int nTiles = Npad >> 4;
  const long tiles = (long)(M >> 4) * nTiles;
  const long wid   = (long)blockIdx.x * 8 + (threadIdx.x >> 5);
  if (wid >= tiles) return;                 // wave-uniform: EXEC stays all-1s

  const int lane = threadIdx.x & 31;
  const int l15  = lane & 15;
  const int hi   = lane >> 4;
  const int nt   = (int)(wid % nTiles);
  const int mt   = (int)(wid / nTiles);
  const int mBase = mt << 4, nBase = nt << 4;

  // A-row coordinates for this lane (ISA A layout: lanes 0-15 & 16-31 both M=row)
  const int m   = mBase + l15;
  const int img = m / HW;
  const int rem = m - img * HW;
  const int yq  = rem / W;
  const int y   = yq, x = rem - yq * W;
  const int pad = KH >> 1;

  // B running pointer: advances by one 32-K chunk (Npad*32 bf16) per step
  const unsigned short* bptr = Wp + ((long)(nBase + l15) << 5) + (hi ? 16 : 0);
  const long bstep = (long)Npad << 5;

  v8f acc = {};

#pragma unroll
  for (int ky = 0; ky < KH; ++ky) {
    const int iy = y + ky - pad;
#pragma unroll
    for (int kx = 0; kx < KW; ++kx) {
      const int ix = x + kx - pad;
      const bool inb = (iy >= 0 && iy < H && ix >= 0 && ix < W);
      const float* p0 = X + (((long)img * H + iy) * W + ix) * (long)Cin
                          + (hi ? 16 : 0);
#pragma unroll 2
      for (int c0 = 0; c0 < Cin; c0 += 32) {
        Frag32B a;
        if (inb) loadA(a, p0 + c0);
        else { a.q[0] = make_uint4(0,0,0,0); a.q[1] = make_uint4(0,0,0,0); }

        Frag32B b;
        const uint4* bp = (const uint4*)bptr;
        b.q[0] = bp[0]; b.q[1] = bp[1];
        __builtin_prefetch(bptr + bstep, 0, 1);   // speculative: safe past end
        bptr += bstep;

        acc = __builtin_amdgcn_wmma_f32_16x16x32_bf16(false, a.v, false, b.v,
                                                      (short)0, acc, false, false);
      }
    }
  }

  // Epilogue: lane's out-channel is constant across its 8 C rows
  const int n  = nBase + l15;
  const bool nv = (n < Nout);
  float sc_ = 1.f, bi_ = 0.f;
  if (nv) { if (scale) sc_ = scale[n]; if (bias) bi_ = bias[n]; }

  // incremental (img,y,x) decode: one div pair total, then +1 with wrap
  int mm   = mBase + (hi << 3);
  int img2 = mm / HW;
  int rem2 = mm - img2 * HW;
  int yy   = rem2 / W;
  int xx   = rem2 - yy * W;

#pragma unroll
  for (int r = 0; r < 8; ++r) {
    float v = acc[r] * sc_ + bi_;
    if (act == 1)      v = v > 0.f ? v : 0.f;
    else if (act == 2) v = 1.f / (1.f + __expf(-v));
    int oy = yy, ox = xx, OH = H, OW = W;
    if (ups) { oy = 2 * yy + ua; ox = 2 * xx + ub; OH = 2 * H; OW = 2 * W; }
    if (nv) {
      long idx;
      if (nchw) idx = (((long)img2 * Nout + n) * OH + oy) * OW + ox;
      else      idx = (((long)img2 * OH + oy) * OW + ox) * (long)Cstride + coff + n;
      Out[idx] = v;
    }
    if (++xx == W) { xx = 0; if (++yy == H) { yy = 0; ++img2; } }
  }
}

// ---------------------------------------------------------------------------
// Weight pack: arbitrary strided f32 source -> WMMA B-layout bf16
// pb[t], t=(kc*Npad+n)*32+kk ; k=kc*32+kk=(ky*KW+kx)*Cin+c
// src value = w[base + n*sN + c*sC + ky*sY + kx*sX], zero-padded for n>=Nout
// ---------------------------------------------------------------------------
__global__ void pack_w_k(const float* __restrict__ w, unsigned short* __restrict__ pb,
                         int Cin, int KH, int KW, int Nout, int Npad,
                         long sN, long sC, long sY, long sX, long base, long total)
{
  long t = (long)blockIdx.x * blockDim.x + threadIdx.x;
  if (t >= total) return;
  const int kk = (int)(t & 31);
  long rest = t >> 5;
  const int n  = (int)(rest % Npad);
  const int kc = (int)(rest / Npad);
  const int k  = (kc << 5) + kk;
  const int c  = k % Cin;
  const int plane = k / Cin;
  const int ky = plane / KW, kx = plane - (plane / KW) * KW;
  float v = (n < Nout) ? w[base + (long)n * sN + (long)c * sC + (long)ky * sY + (long)kx * sX]
                       : 0.f;
  pb[t] = f2bf(v);
}

// ---------------------------------------------------------------------------
// Small helpers
// ---------------------------------------------------------------------------
__global__ void bnfold_k(const float* g, const float* b, const float* m,
                         const float* v, float* sc, float* bi, int C)
{
  int c = blockIdx.x * blockDim.x + threadIdx.x;
  if (c < C) { float s = g[c] * rsqrtf(v[c] + 1e-5f); sc[c] = s; bi[c] = b[c] - m[c] * s; }
}

__global__ void colsq_k(const float* __restrict__ E, float* __restrict__ E2)
{
  int e = blockIdx.x * blockDim.x + threadIdx.x;
  if (e < 512) { float s = 0.f;
    for (int d = 0; d < 64; ++d) { float v = E[d * 512 + e]; s += v * v; }
    E2[e] = s; }
}

__global__ void nchw2nhwc_k(const float* __restrict__ in, float* __restrict__ out,
                            int N, int C, int H, int W, int Cstride, int coff)
{
  long idx = (long)blockIdx.x * blockDim.x + threadIdx.x;
  long total = (long)N * C * H * W;
  if (idx >= total) return;
  int xx = (int)(idx % W); long t2 = idx / W;
  int yy = (int)(t2 % H);  long t3 = t2 / H;
  int c  = (int)(t3 % C);  int n = (int)(t3 / C);
  out[(((long)n * H + yy) * W + xx) * Cstride + coff + c] = in[idx];
}

__global__ void copych_k(const float* __restrict__ in, float* __restrict__ out,
                         long NHW, int C, int Ctot, int coff)
{
  long idx = (long)blockIdx.x * blockDim.x + threadIdx.x;
  if (idx >= NHW * C) return;
  int c = (int)(idx % C); long p = idx / C;
  out[p * (long)Ctot + coff + c] = in[idx];
}

__global__ void maxpool2_k(const float* __restrict__ in, float* __restrict__ out,
                           int N, int H, int W, int C)
{
  long idx = (long)blockIdx.x * blockDim.x + threadIdx.x;
  const int H2 = H >> 1, W2 = W >> 1;
  long total = (long)N * H2 * W2 * C;
  if (idx >= total) return;
  int c = (int)(idx % C); long p = idx / C;
  int xx = (int)(p % W2); p /= W2;
  int yy = (int)(p % H2); int n = (int)(p / H2);
  long b0 = (((long)n * H + 2 * yy) * W + 2 * xx) * C + c;
  float a = in[b0], b = in[b0 + C], cc = in[b0 + (long)W * C], d = in[b0 + (long)W * C + C];
  out[idx] = fmaxf(fmaxf(a, b), fmaxf(cc, d));
}

// argmin over 512 codes of (E2[e]-2*XE[p,e]); write q (to ws + e-output);
// accumulate mean((q-h)^2) into diff slot.
__global__ void vq_argmin_k(const float* __restrict__ XE, const float* __restrict__ E2,
                            const float* __restrict__ E,  const float* __restrict__ Hf,
                            float* __restrict__ qws, float* __restrict__ qout,
                            float* __restrict__ diffslot, int Npts, float invTot)
{
  int p = blockIdx.x * blockDim.x + threadIdx.x;
  if (p >= Npts) return;
  const float* row = XE + (long)p * 512;
  float best = 3.4e38f; int bi = 0;
  for (int e = 0; e < 512; ++e) {
    float d = E2[e] - 2.f * row[e];
    if (d < best) { best = d; bi = e; }
  }
  float local = 0.f;
  for (int d = 0; d < 64; ++d) {
    float q = E[d * 512 + bi];
    qws[(long)p * 64 + d]  = q;
    qout[(long)p * 64 + d] = q;
    float df = q - Hf[(long)p * 64 + d];
    local += df * df;
  }
  atomicAdd(diffslot, local * invTot);
}

__global__ void init_diff_k(float* dst, const float* src) { dst[0] = src[0]; }

// ---------------------------------------------------------------------------
// Host orchestration
// ---------------------------------------------------------------------------
static inline void gemm(hipStream_t s, const float* X, const unsigned short* Wp,
                        const float* scale, const float* bias, float* Out,
                        int Nimg, int H, int W, int Cin, int KH,
                        int Nout, int Npad, int ups, int ua, int ub,
                        int Cstride, int coff, int nchw, int act)
{
  long M = (long)Nimg * H * W;
  long tiles = (M >> 4) * (Npad >> 4);
  int grid = (int)((tiles + 7) / 8);
  if (KH == 3)
    conv_wmma_k<3,3><<<grid, 256, 0, s>>>(X, Wp, scale, bias, Out, Nimg, H, W, Cin,
                                          Nout, Npad, ups, ua, ub, Cstride, coff, nchw, act);
  else
    conv_wmma_k<1,1><<<grid, 256, 0, s>>>(X, Wp, scale, bias, Out, Nimg, H, W, Cin,
                                          Nout, Npad, ups, ua, ub, Cstride, coff, nchw, act);
}

static inline void pack(hipStream_t s, const float* w, unsigned short* pb,
                        int Cin, int KH, int Nout, int Npad,
                        long sN, long sC, long sY, long sX, long base)
{
  long total = (long)KH * KH * Cin * Npad;
  int grid = (int)((total + 255) / 256);
  pack_w_k<<<grid, 256, 0, s>>>(w, pb, Cin, KH, KH, Nout, Npad, sN, sC, sY, sX, base, total);
}

extern "C" void kernel_launch(void* const* d_in, const int* in_sizes, int n_in,
                              void* d_out, int out_size, void* d_ws, size_t ws_size,
                              hipStream_t stream)
{
  (void)in_sizes; (void)n_in; (void)out_size; (void)ws_size;
  auto F = [&](int i) { return (const float*)d_in[i]; };

  // ---- d_in order: JAX pytree (sorted-key) flatten of setup_inputs() ----
  int t = 0;
  const float* in_diff = F(t++);        // diff_1
  const float* in_ei   = F(t++);        // embed_ind_1 [8,128,128,64] BHWC
  struct DC { const float *b1,*b2,*g1,*g2,*m1,*m2,*v1,*v2,*w1,*w2; } down[3];
  for (int i = 0; i < 3; ++i) {
    down[i].b1=F(t++); down[i].b2=F(t++); down[i].g1=F(t++); down[i].g2=F(t++);
    down[i].m1=F(t++); down[i].m2=F(t++); down[i].v1=F(t++); down[i].v2=F(t++);
    down[i].w1=F(t++); down[i].w2=F(t++);
  }
  const float* outc_b = F(t++); const float* outc_w = F(t++);
  struct UP { const float *b1,*b2,*g1,*g2,*m1,*m2,*tb,*tw,*v1,*v2,*w1,*w2; } up[3];
  for (int i = 0; i < 3; ++i) {
    up[i].b1=F(t++); up[i].b2=F(t++); up[i].g1=F(t++); up[i].g2=F(t++);
    up[i].m1=F(t++); up[i].m2=F(t++); up[i].tb=F(t++); up[i].tw=F(t++);
    up[i].v1=F(t++); up[i].v2=F(t++); up[i].w1=F(t++); up[i].w2=F(t++);
  }
  struct VQ { const float *dec_b,*dec_w,*embed,*enc_b,*enc_w; } vq[3];
  for (int i = 0; i < 3; ++i) {
    vq[i].dec_b=F(t++); vq[i].dec_w=F(t++); vq[i].embed=F(t++);
    vq[i].enc_b=F(t++); vq[i].enc_w=F(t++);
  }
  const float* qx1 = F(t++);
  const float* x1  = F(t++);

  float* out = (float*)d_out;
  const long OUT_DIFF = 393216, OUT_EI = 393217;
  const long OUT_E2 = OUT_EI + 8388608, OUT_E3 = OUT_E2 + 2097152, OUT_E4 = OUT_E3 + 524288;

  // ---- workspace arena ----
  char* ws = (char*)d_ws; size_t off = 0;
  auto allocB = [&](size_t bytes) -> void* {
    off = (off + 255) & ~(size_t)255; void* p = ws + off; off += bytes; return p; };
  auto aF = [&](size_t e) { return (float*)allocB(e * 4); };
  auto aU = [&](size_t e) { return (unsigned short*)allocB(e * 2); };

  float* bufA = aF(8388608);  // x1 NHWC ; reused as up2 concat [8,64,64,256]
  float* pool = aF(2097152);  // pooled1/2/3
  float* d1a  = aF(4194304);  // down1 mid ; reused as u2a
  float* x2b  = aF(4194304);  // x2 (pre & post VQ)
  float* x3b  = aF(2097152);
  float* x4b  = aF(1048576);
  float* d2a  = aF(2097152);  // down2 mid ; reused as u1a
  float* d3a  = aF(1048576);
  float* hbuf = aF(2097152);  // VQ enc outputs (max 32768x64)
  float* xe   = aF(16777216); // XE [N,512] ; reused as up3 concat [8,128,128,128]
  float* qws  = aF(2097152);  // aligned q for decoder input
  float* c1   = aF(4194304);  // up1 concat [8,32,32,512] ; reused as u2b
  float* u1b  = aF(2097152);
  float* u3a  = aF(8388608);
  float* u3b  = aF(8388608);
  float* e2buf = aF(512);

  // fold BN into per-channel scale/bias for the 12 convs
  struct CL { const float *w,*g,*b,*m,*v; int Cin, Cout; } cl[12] = {
    {down[0].w1, down[0].g1, down[0].b1, down[0].m1, down[0].v1,  64, 128},
    {down[0].w2, down[0].g2, down[0].b2, down[0].m2, down[0].v2, 128, 128},
    {down[1].w1, down[1].g1, down[1].b1, down[1].m1, down[1].v1, 128, 256},
    {down[1].w2, down[1].g2, down[1].b2, down[1].m2, down[1].v2, 256, 256},
    {down[2].w1, down[2].g1, down[2].b1, down[2].m1, down[2].v1, 256, 512},
    {down[2].w2, down[2].g2, down[2].b2, down[2].m2, down[2].v2, 512, 512},
    {up[0].w1, up[0].g1, up[0].b1, up[0].m1, up[0].v1, 512, 256},
    {up[0].w2, up[0].g2, up[0].b2, up[0].m2, up[0].v2, 256, 256},
    {up[1].w1, up[1].g1, up[1].b1, up[1].m1, up[1].v1, 256, 128},
    {up[1].w2, up[1].g2, up[1].b2, up[1].m2, up[1].v2, 128, 128},
    {up[2].w1, up[2].g1, up[2].b1, up[2].m1, up[2].v1, 128,  64},
    {up[2].w2, up[2].g2, up[2].b2, up[2].m2, up[2].v2,  64,  64},
  };
  unsigned short* pk[12]; float* scb[12]; float* bib[12];
  for (int i = 0; i < 12; ++i) {
    pk[i]  = aU((size_t)cl[i].Cin * 9 * cl[i].Cout);
    scb[i] = aF(cl[i].Cout); bib[i] = aF(cl[i].Cout);
    bnfold_k<<<(cl[i].Cout + 255) / 256, 256, 0, stream>>>(cl[i].g, cl[i].b, cl[i].m,
                                                           cl[i].v, scb[i], bib[i], cl[i].Cout);
    pack(stream, cl[i].w, pk[i], cl[i].Cin, 3, cl[i].Cout, cl[i].Cout,
         (long)cl[i].Cin * 9, 9, 3, 1, 0);
  }
  // outc (3->pad16), VQ enc/dec/embed, transpose-conv slices (kernel flipped)
  unsigned short* pk_outc = aU(64 * 16);
  pack(stream, outc_w, pk_outc, 64, 1, 3, 16, 64, 1, 0, 0, 0);
  const int vqC[3] = {128, 256, 512};
  unsigned short *pk_enc[3], *pk_dec[3], *pk_emb[3], *pk_tw[3][4];
  for (int i = 0; i < 3; ++i) {
    pk_enc[i] = aU((size_t)vqC[i] * 64);
    pk_dec[i] = aU((size_t)64 * vqC[i]);
    pk_emb[i] = aU((size_t)64 * 512);
    pack(stream, vq[i].enc_w, pk_enc[i], vqC[i], 1, 64, 64, vqC[i], 1, 0, 0, 0);
    pack(stream, vq[i].dec_w, pk_dec[i], 64, 1, vqC[i], vqC[i], 64, 1, 0, 0, 0);
    pack(stream, vq[i].embed, pk_emb[i], 64, 1, 512, 512, 1, 512, 0, 0, 0);
  }
  const int twI[3] = {512, 256, 128};            // transpose-conv in channels
  for (int i = 0; i < 3; ++i)
    for (int a = 0; a < 2; ++a)
      for (int b = 0; b < 2; ++b) {
        pk_tw[i][a * 2 + b] = aU((size_t)twI[i] * (twI[i] / 2));
        pack(stream, up[i].tw, pk_tw[i][a * 2 + b], twI[i], 1, twI[i] / 2, twI[i] / 2,
             (long)twI[i] * 4, 4, 0, 0, (long)(1 - a) * 2 + (1 - b));
      }

  // ---- outputs that are pure passthrough / init ----
  init_diff_k<<<1, 1, 0, stream>>>(out + OUT_DIFF, in_diff);
  hipMemcpyAsync(out + OUT_EI, in_ei, (size_t)8388608 * 4,
                 hipMemcpyDeviceToDevice, stream);

  // ---- encoder path ----
  nchw2nhwc_k<<<(8388608 + 255) / 256, 256, 0, stream>>>(x1, bufA, 8, 64, 128, 128, 64, 0);
  maxpool2_k<<<(2097152 + 255) / 256, 256, 0, stream>>>(bufA, pool, 8, 128, 128, 64);
  gemm(stream, pool, pk[0], scb[0], bib[0], d1a, 8, 64, 64,  64, 3, 128, 128, 0,0,0, 128, 0, 0, 1);
  gemm(stream, d1a,  pk[1], scb[1], bib[1], x2b, 8, 64, 64, 128, 3, 128, 128, 0,0,0, 128, 0, 0, 1);

  const int vqH[3] = {64, 32, 16};
  const int vqN[3] = {32768, 8192, 2048};
  float* vqX[3] = {x2b, x3b, x4b};
  const long vqOff[3] = {OUT_E2, OUT_E3, OUT_E4};
  auto run_vq = [&](int i) {
    colsq_k<<<2, 256, 0, stream>>>(vq[i].embed, e2buf);
    gemm(stream, vqX[i], pk_enc[i], nullptr, vq[i].enc_b, hbuf,
         8, vqH[i], vqH[i], vqC[i], 1, 64, 64, 0,0,0, 64, 0, 0, 0);
    gemm(stream, hbuf, pk_emb[i], nullptr, nullptr, xe,
         8, vqH[i], vqH[i], 64, 1, 512, 512, 0,0,0, 512, 0, 0, 0);
    vq_argmin_k<<<(vqN[i] + 255) / 256, 256, 0, stream>>>(
        xe, e2buf, vq[i].embed, hbuf, qws, out + vqOff[i], out + OUT_DIFF,
        vqN[i], 1.f / (float)(vqN[i] * 64));
    gemm(stream, qws, pk_dec[i], nullptr, vq[i].dec_b, vqX[i],
         8, vqH[i], vqH[i], 64, 1, vqC[i], vqC[i], 0,0,0, vqC[i], 0, 0, 0);
  };

  run_vq(0);
  maxpool2_k<<<(1048576 + 255) / 256, 256, 0, stream>>>(x2b, pool, 8, 64, 64, 128);
  gemm(stream, pool, pk[2], scb[2], bib[2], d2a, 8, 32, 32, 128, 3, 256, 256, 0,0,0, 256, 0, 0, 1);
  gemm(stream, d2a,  pk[3], scb[3], bib[3], x3b, 8, 32, 32, 256, 3, 256, 256, 0,0,0, 256, 0, 0, 1);
  run_vq(1);
  maxpool2_k<<<(524288 + 255) / 256, 256, 0, stream>>>(x3b, pool, 8, 32, 32, 256);
  gemm(stream, pool, pk[4], scb[4], bib[4], d3a, 8, 16, 16, 256, 3, 512, 512, 0,0,0, 512, 0, 0, 1);
  gemm(stream, d3a,  pk[5], scb[5], bib[5], x4b, 8, 16, 16, 512, 3, 512, 512, 0,0,0, 512, 0, 0, 1);
  run_vq(2);

  // ---- up1: concat [x3q | upsample(x4q)] -> double conv ----
  copych_k<<<(2097152 + 255) / 256, 256, 0, stream>>>(x3b, c1, 8192, 256, 512, 0);
  for (int a = 0; a < 2; ++a) for (int b = 0; b < 2; ++b)
    gemm(stream, x4b, pk_tw[0][a * 2 + b], nullptr, up[0].tb, c1,
         8, 16, 16, 512, 1, 256, 256, 1, a, b, 512, 256, 0, 0);
  float* u1a = d2a;
  gemm(stream, c1,  pk[6], scb[6], bib[6], u1a, 8, 32, 32, 512, 3, 256, 256, 0,0,0, 256, 0, 0, 1);
  gemm(stream, u1a, pk[7], scb[7], bib[7], u1b, 8, 32, 32, 256, 3, 256, 256, 0,0,0, 256, 0, 0, 1);

  // ---- up2: concat [x2q | upsample(u1b)] ----
  float* c2 = bufA;
  copych_k<<<(4194304 + 255) / 256, 256, 0, stream>>>(x2b, c2, 32768, 128, 256, 0);
  for (int a = 0; a < 2; ++a) for (int b = 0; b < 2; ++b)
    gemm(stream, u1b, pk_tw[1][a * 2 + b], nullptr, up[1].tb, c2,
         8, 32, 32, 256, 1, 128, 128, 1, a, b, 256, 128, 0, 0);
  float* u2a = d1a; float* u2b = c1;
  gemm(stream, c2,  pk[8], scb[8], bib[8], u2a, 8, 64, 64, 256, 3, 128, 128, 0,0,0, 128, 0, 0, 1);
  gemm(stream, u2a, pk[9], scb[9], bib[9], u2b, 8, 64, 64, 128, 3, 128, 128, 0,0,0, 128, 0, 0, 1);

  // ---- up3: concat [qx1 | upsample(u2b)] ----
  float* c3 = xe;
  nchw2nhwc_k<<<(8388608 + 255) / 256, 256, 0, stream>>>(qx1, c3, 8, 64, 128, 128, 128, 0);
  for (int a = 0; a < 2; ++a) for (int b = 0; b < 2; ++b)
    gemm(stream, u2b, pk_tw[2][a * 2 + b], nullptr, up[2].tb, c3,
         8, 64, 64, 128, 1, 64, 64, 1, a, b, 128, 64, 0, 0);
  gemm(stream, c3,  pk[10], scb[10], bib[10], u3a, 8, 128, 128, 128, 3, 64, 64, 0,0,0, 64, 0, 0, 1);
  gemm(stream, u3a, pk[11], scb[11], bib[11], u3b, 8, 128, 128,  64, 3, 64, 64, 0,0,0, 64, 0, 0, 1);

  // ---- outc: 1x1 -> sigmoid -> NCHW directly into d_out ----
  gemm(stream, u3b, pk_outc, nullptr, outc_b, out, 8, 128, 128, 64, 1, 3, 16,
       0, 0, 0, 0, 0, 1, 2);
}
